// WindowAttention_64012192579819
// MI455X (gfx1250) — matrix operand order repaired and checked by reading, not compiled, not run on hardware
//
#include <hip/hip_runtime.h>

typedef unsigned short u16;
typedef __attribute__((ext_vector_type(16))) __bf16 v16bf;
typedef __attribute__((ext_vector_type(8)))  float  v8f;

#define SCALE_Q 0.17677669529663687f   // 32^-0.5

// ---------- helpers ----------
__device__ __forceinline__ u16 f2bf(float f) {
  union { float f; unsigned u; } x; x.f = f;
  unsigned u = x.u;
  return (u16)((u + 0x7FFFu + ((u >> 16) & 1u)) >> 16);
}

// A fragment (16x32 bf16, row-major, ld in elements):
// lane l: m = l&15 ; VGPR0-3 = K[ko..ko+7], VGPR4-7 = K[16+ko..16+ko+7], ko = (l>=16)?8:0
__device__ __forceinline__ v16bf load_a(const u16* base, int ld) {
  int l  = threadIdx.x & 31;
  int m  = l & 15;
  int ko = (l >> 4) * 8;
  const u16* p = base + (size_t)m * ld + ko;
  union { uint4 q[2]; v16bf v; } u;
  u.q[0] = *(const uint4*)(p);
  u.q[1] = *(const uint4*)(p + 16);
  return u.v;
}

// B fragment from B^T stored [N][K] row-major (ld in elements):
// lane l: n = l&15 ; elements 0-7 = K[ko..ko+7], 8-15 = K[16+ko..16+ko+7], ko = (l>=16)?8:0
__device__ __forceinline__ v16bf load_b(const u16* baseT, int ld) {
  int l  = threadIdx.x & 31;
  int n  = l & 15;
  int ko = (l >> 4) * 8;
  const u16* p = baseT + (size_t)n * ld + ko;
  union { uint4 q[2]; v16bf v; } u;
  u.q[0] = *(const uint4*)(p);
  u.q[1] = *(const uint4*)(p + 16);
  return u.v;
}

__device__ __forceinline__ v8f wmma_bf(v16bf a, v16bf b, v8f c) {
  return __builtin_amdgcn_wmma_f32_16x16x32_bf16(false, a, false, b, (short)0, c, false, false);
}

// ---------- conversion ----------
__global__ void __launch_bounds__(256) k_cvt(const float* __restrict__ in,
                                             u16* __restrict__ out, int n) {
  for (int i = blockIdx.x * 256 + threadIdx.x; i < n; i += gridDim.x * 256)
    out[i] = f2bf(in[i]);
}

// zero pad rows (49..63) of q,k and pad cols of vT (per b*h)
__global__ void __launch_bounds__(512) k_zero_pads(u16* q, u16* k, u16* vT) {
  int bh = blockIdx.x;           // 0..8191
  int t = threadIdx.x;
  if (t < 480) {
    int r = 49 + t / 32, d = t % 32;
    q[((size_t)bh * 64 + r) * 32 + d] = 0;
    k[((size_t)bh * 64 + r) * 32 + d] = 0;
    int d2 = t / 15, c = 49 + t % 15;
    vT[((size_t)bh * 32 + d2) * 64 + c] = 0;
  }
}

// ---------- image QKV GEMM: [50176,768] = xb[50176,256] @ Wqkv^T ----------
// Async-DMA double-buffered LDS staging: per k-step stage A[128][32] + B[64][32]
// via global_load_async_to_lds_b128 (3 x 16B per thread), synchronized with
// s_wait_asynccnt (async loads complete in-order) + workgroup barrier.
__global__ void __launch_bounds__(256) k_qkv_img(const u16* __restrict__ xb,
                                                 const u16* __restrict__ Wb,
                                                 const float* __restrict__ bias,
                                                 u16* q, u16* k, u16* vT) {
  __shared__ __align__(16) u16 sA[2][128][32];
  __shared__ __align__(16) u16 sB[2][64][32];
  int t = threadIdx.x;
  int mblk = blockIdx.x / 12, nblk = blockIdx.x % 12;
  int wave = t >> 5, l = t & 31;
  int m0 = mblk * 128, n0 = nblk * 64;

  // per-thread 16B copy slots: A chunks t and t+256 (512 total), B chunk t (256 total)
  int crow = t >> 2, ccol = (t & 3) * 8;   // chunk row, chunk col (halfs)
  unsigned long long gA0 = (unsigned long long)(uintptr_t)xb +
                           ((size_t)(m0 + crow) * 256 + ccol) * 2;
  unsigned long long gA1 = gA0 + (size_t)64 * 256 * 2;
  unsigned long long gB  = (unsigned long long)(uintptr_t)Wb +
                           ((size_t)(n0 + crow) * 256 + ccol) * 2;
  unsigned ldsA0[2], ldsA1[2], ldsB[2];
#pragma unroll
  for (int f = 0; f < 2; ++f) {
    ldsA0[f] = (unsigned)(uintptr_t)&sA[f][crow][ccol];
    ldsA1[f] = (unsigned)(uintptr_t)&sA[f][64 + crow][ccol];
    ldsB[f]  = (unsigned)(uintptr_t)&sB[f][crow][ccol];
  }
  auto issue = [&](int stage) {
    int f = stage & 1;
    unsigned long long off = (unsigned long long)stage * 64;  // 32 halfs per k-step
    asm volatile("global_load_async_to_lds_b128 %0, %1, off"
                 :: "v"(ldsA0[f]), "v"(gA0 + off) : "memory");
    asm volatile("global_load_async_to_lds_b128 %0, %1, off"
                 :: "v"(ldsA1[f]), "v"(gA1 + off) : "memory");
    asm volatile("global_load_async_to_lds_b128 %0, %1, off"
                 :: "v"(ldsB[f]),  "v"(gB  + off) : "memory");
  };

  issue(0);
  v8f acc[2][2] = {};
  int wm = (wave >> 1) * 32, wn = (wave & 1) * 32;
  for (int s = 0; s < 8; ++s) {
    int cur = s & 1;
    if (s < 7) {
      issue(s + 1);
      asm volatile("s_wait_asynccnt 0x3" ::: "memory");  // oldest 3 (stage s) done
    } else {
      asm volatile("s_wait_asynccnt 0x0" ::: "memory");
    }
    __syncthreads();
    v16bf a0 = load_a(&sA[cur][wm][0], 32);
    v16bf a1 = load_a(&sA[cur][wm + 16][0], 32);
    v16bf b0 = load_b(&sB[cur][wn][0], 32);
    v16bf b1 = load_b(&sB[cur][wn + 16][0], 32);
    acc[0][0] = wmma_bf(a0, b0, acc[0][0]);
    acc[0][1] = wmma_bf(a0, b1, acc[0][1]);
    acc[1][0] = wmma_bf(a1, b0, acc[1][0]);
    acc[1][1] = wmma_bf(a1, b1, acc[1][1]);
    __syncthreads();
  }

  int nlo = l & 15, mhi = (l >> 4) * 8;
  int wm0 = m0 + wm, wn0 = n0 + wn;
  for (int mt = 0; mt < 2; ++mt)
    for (int nt = 0; nt < 2; ++nt)
#pragma unroll
      for (int r = 0; r < 8; ++r) {
        int tok = wm0 + mt * 16 + mhi + r;
        int f = wn0 + nt * 16 + nlo;
        float val = acc[mt][nt][r] + bias[f];
        int sel = f >> 8, h = (f >> 5) & 7, d = f & 31;
        int b = tok / 49, pos = tok % 49;
        size_t bh = (size_t)b * 8 + h;
        if (sel == 0)      q[(bh * 64 + pos) * 32 + d] = f2bf(val * SCALE_Q);
        else if (sel == 1) k[(bh * 64 + pos) * 32 + d] = f2bf(val);
        else               vT[(bh * 32 + d) * 64 + pos] = f2bf(val);
      }
}

// ---------- text QKV GEMM: [1024,768] = xtb[1024,512] @ Wqkv_t^T ----------
__global__ void __launch_bounds__(256) k_qkv_txt(const u16* __restrict__ xb,
                                                 const u16* __restrict__ Wb,
                                                 const float* __restrict__ bias,
                                                 u16* qt, u16* kt, u16* vtT) {
  const int K = 512;
  int mblk = blockIdx.x / 12, nblk = blockIdx.x % 12;
  int wave = threadIdx.x >> 5, l = threadIdx.x & 31;
  int m0 = mblk * 128 + (wave >> 1) * 32;
  int n0 = nblk * 64 + (wave & 1) * 32;
  v8f acc[2][2] = {};
  for (int kk = 0; kk < K; kk += 32) {
    v16bf a0 = load_a(xb + (size_t)m0 * K + kk, K);
    v16bf a1 = load_a(xb + (size_t)(m0 + 16) * K + kk, K);
    v16bf b0 = load_b(Wb + (size_t)n0 * K + kk, K);
    v16bf b1 = load_b(Wb + (size_t)(n0 + 16) * K + kk, K);
    acc[0][0] = wmma_bf(a0, b0, acc[0][0]);
    acc[0][1] = wmma_bf(a0, b1, acc[0][1]);
    acc[1][0] = wmma_bf(a1, b0, acc[1][0]);
    acc[1][1] = wmma_bf(a1, b1, acc[1][1]);
  }
  int nlo = l & 15, mhi = (l >> 4) * 8;
  for (int mt = 0; mt < 2; ++mt)
    for (int nt = 0; nt < 2; ++nt)
#pragma unroll
      for (int r = 0; r < 8; ++r) {
        int tok = m0 + mt * 16 + mhi + r;
        int f = n0 + nt * 16 + nlo;
        float val = acc[mt][nt][r] + bias[f];
        int sel = f >> 8, h = (f >> 5) & 7, d = f & 31;
        int bt = tok >> 6, tt = tok & 63;
        size_t bh = (size_t)bt * 8 + h;
        if (sel == 0)      qt[(bh * 64 + tt) * 32 + d] = f2bf(val * SCALE_Q);
        else if (sel == 1) kt[(bh * 64 + tt) * 32 + d] = f2bf(val);
        else               vtT[(bh * 32 + d) * 64 + tt] = f2bf(val);
      }
}

// ---------- fused image attention: one block per (b,h) ----------
__global__ void __launch_bounds__(256) k_attn_img(const u16* __restrict__ q,
                                                  const u16* __restrict__ k,
                                                  const u16* __restrict__ vT,
                                                  const u16* __restrict__ kt,
                                                  const u16* __restrict__ vtT,
                                                  const float* __restrict__ rpb,
                                                  const float* __restrict__ mask,
                                                  const float* __restrict__ mask_text,
                                                  const float* __restrict__ i2t,
                                                  u16* __restrict__ attn_out) {
  __shared__ __align__(16) float sS[64][128];
  __shared__ __align__(16) u16   sP[64][128];
  int b = blockIdx.x >> 3, h = blockIdx.x & 7;
  int w = b & 63, bt = b >> 6;
  size_t bh = (size_t)b * 8 + h, bth = (size_t)bt * 8 + h;
  int wave = threadIdx.x >> 5, l = threadIdx.x & 31;
  int nlo = l & 15, mhi = (l >> 4) * 8;

  // phase 1: logits -> LDS.  32 tiles (4m x 8n); cols 0-63 image keys, 64-127 text keys
  for (int i = 0; i < 4; ++i) {
    int tile = wave * 4 + i;
    int mt = tile >> 3, nt = tile & 7;
    v16bf a = load_a(q + (bh * 64 + mt * 16) * 32, 32);
    v16bf bf = (nt < 4) ? load_b(k + (bh * 64 + nt * 16) * 32, 32)
                        : load_b(kt + (bth * 64 + (nt - 4) * 16) * 32, 32);
    v8f z = {};
    v8f s = wmma_bf(a, bf, z);
#pragma unroll
    for (int r = 0; r < 8; ++r)
      sS[mt * 16 + mhi + r][nt * 16 + nlo] = s[r];
  }
  __syncthreads();

  // phase 2: softmax rows (+rpb, +shift mask, +text mask/bias), probs -> LDS bf16
  if (threadIdx.x < 64) {
    int row = threadIdx.x;
    if (row < 49) {
      int nh = row / 7, nw = row % 7;
      float mx = -1e30f;
      for (int m = 0; m < 49; ++m) {
        int mh = m / 7, mw = m % 7;
        int idx = (nh - mh + 6) * 13 + (nw - mw + 6);
        float v = sS[row][m] + rpb[idx * 8 + h] + mask[((size_t)w * 49 + row) * 49 + m];
        sS[row][m] = v; mx = fmaxf(mx, v);
      }
      for (int s = 0; s < 64; ++s) {
        float v = sS[row][64 + s] + mask_text[bt * 64 + s] + i2t[h * 64 + s];
        sS[row][64 + s] = v; mx = fmaxf(mx, v);
      }
      float sum = 0.f;
      for (int m = 0; m < 49; ++m) { float e = __expf(sS[row][m] - mx); sS[row][m] = e; sum += e; }
      for (int s = 0; s < 64; ++s) { float e = __expf(sS[row][64 + s] - mx); sS[row][64 + s] = e; sum += e; }
      float inv = 1.f / sum;
      for (int m = 0; m < 49; ++m) sP[row][m] = f2bf(sS[row][m] * inv);
      for (int c = 49; c < 64; ++c) sP[row][c] = 0;
      for (int s = 0; s < 64; ++s)  sP[row][64 + s] = f2bf(sS[row][64 + s] * inv);
    } else {
      for (int c = 0; c < 128; ++c) sP[row][c] = 0;
    }
  }
  __syncthreads();

  // phase 3: PV. 8 tiles (4m x 2n of hd), K chunks: 2 image (vT) + 2 text (vtT)
  {
    int mt = wave >> 1, nt = wave & 1;
    v8f acc = {};
#pragma unroll
    for (int c = 0; c < 4; ++c) {
      v16bf a = load_a(&sP[mt * 16][c * 32], 128);
      v16bf bf = (c < 2) ? load_b(vT + (bh * 32 + nt * 16) * 64 + c * 32, 64)
                         : load_b(vtT + (bth * 32 + nt * 16) * 64 + (c - 2) * 32, 64);
      acc = wmma_bf(a, bf, acc);
    }
#pragma unroll
    for (int r = 0; r < 8; ++r) {
      int pos = mt * 16 + mhi + r;
      if (pos < 49) {
        int d = nt * 16 + nlo;
        attn_out[((size_t)b * 49 + pos) * 256 + h * 32 + d] = f2bf(acc[r]);
      }
    }
  }
}

// ---------- flash-style text attention: one block (4 waves) per (bt,h) ----------
__global__ void __launch_bounds__(128) k_attn_text(const u16* __restrict__ qt,
                                                   const u16* __restrict__ k,
                                                   const u16* __restrict__ vT,
                                                   const u16* __restrict__ kt,
                                                   const u16* __restrict__ vtT,
                                                   const float* __restrict__ i2t,
                                                   const float* __restrict__ t2t,
                                                   const float* __restrict__ mask_text,
                                                   u16* __restrict__ out_text) {
  __shared__ __align__(16) u16 sP[4][16][64];
  int bt = blockIdx.x >> 3, h = blockIdx.x & 7;
  size_t bth = (size_t)bt * 8 + h;
  int wv = threadIdx.x >> 5, l = threadIdx.x & 31;
  int nlo = l & 15, mhi = (l >> 4) * 8;
  int m0 = wv * 16;

  v16bf a_q = load_a(qt + (bth * 64 + m0) * 32, 32);
  v8f o0 = {}, o1 = {};
  float rmax[8], rsum[8];
#pragma unroll
  for (int r = 0; r < 8; ++r) { rmax[r] = -1e30f; rsum[r] = 0.f; }

  for (int kb = 0; kb < 65; ++kb) {
    bool is_text = (kb == 64);
    const u16* kbase; const u16* vbase;
    if (!is_text) {
      size_t bhh = ((size_t)bt * 64 + kb) * 8 + h;
      kbase = k + bhh * 64 * 32;
      vbase = vT + bhh * 32 * 64;
    } else {
      kbase = kt + bth * 64 * 32;
      vbase = vtT + bth * 32 * 64;
    }
    // S = q_t * K^T  (4 n-tiles of 16 keys)
    v8f s[4];
#pragma unroll
    for (int j = 0; j < 4; ++j) {
      v16bf bf = load_b(kbase + (size_t)j * 16 * 32, 32);
      v8f z = {};
      s[j] = wmma_bf(a_q, bf, z);
    }
    // biases / masks
#pragma unroll
    for (int j = 0; j < 4; ++j)
#pragma unroll
      for (int r = 0; r < 8; ++r) {
        int col = j * 16 + nlo;
        int trow = m0 + mhi + r;
        float v = s[j][r];
        if (!is_text) {
          v += i2t[(8 + h) * 64 + trow];
          if (col >= 49) v = -1e30f;
        } else {
          v += t2t[((size_t)h * 64 + trow) * 64 + col] + mask_text[bt * 64 + col];
        }
        s[j][r] = v;
      }
    // online softmax update per row
#pragma unroll
    for (int r = 0; r < 8; ++r) {
      float bm = fmaxf(fmaxf(s[0][r], s[1][r]), fmaxf(s[2][r], s[3][r]));
      for (int off = 1; off < 16; off <<= 1) bm = fmaxf(bm, __shfl_xor(bm, off, 32));
      float nm = fmaxf(rmax[r], bm);
      float sc = __expf(rmax[r] - nm);
      rmax[r] = nm;
      o0[r] *= sc; o1[r] *= sc;
      float ps = 0.f;
#pragma unroll
      for (int j = 0; j < 4; ++j) { float e = __expf(s[j][r] - nm); s[j][r] = e; ps += e; }
      for (int off = 1; off < 16; off <<= 1) ps += __shfl_xor(ps, off, 32);
      rsum[r] = rsum[r] * sc + ps;
    }
    // stage probs (C layout) to LDS, reload as A fragments, PV
    __syncthreads();
#pragma unroll
    for (int j = 0; j < 4; ++j)
#pragma unroll
      for (int r = 0; r < 8; ++r)
        sP[wv][mhi + r][j * 16 + nlo] = f2bf(s[j][r]);
    __syncthreads();
#pragma unroll
    for (int kc = 0; kc < 2; ++kc) {
      v16bf ap = load_a(&sP[wv][0][kc * 32], 64);
      v16bf b0 = load_b(vbase + kc * 32, 64);
      v16bf b1 = load_b(vbase + 16 * 64 + kc * 32, 64);
      o0 = wmma_bf(ap, b0, o0);
      o1 = wmma_bf(ap, b1, o1);
    }
  }
  // normalize + store [bt*64+t][h*32+d]
#pragma unroll
  for (int r = 0; r < 8; ++r) {
    float inv = 1.f / rsum[r];
    int trow = m0 + mhi + r;
    size_t base = ((size_t)bt * 64 + trow) * 256 + h * 32;
    out_text[base + nlo]      = f2bf(o0[r] * inv);
    out_text[base + 16 + nlo] = f2bf(o1[r] * inv);
  }
}

// ---------- projection GEMM (fp32 out + bias): [M,N] = A[M,K] @ W[N,K]^T ----------
__global__ void __launch_bounds__(256) k_proj(const u16* __restrict__ A,
                                              const u16* __restrict__ Wb,
                                              const float* __restrict__ bias,
                                              float* __restrict__ out,
                                              int N, int K, int nbn) {
  int mblk = blockIdx.x / nbn, nblk = blockIdx.x % nbn;
  int wave = threadIdx.x >> 5, l = threadIdx.x & 31;
  int m0 = mblk * 128 + (wave >> 1) * 32;
  int n0 = nblk * 64 + (wave & 1) * 32;
  v8f acc[2][2] = {};
  for (int kk = 0; kk < K; kk += 32) {
    v16bf a0 = load_a(A + (size_t)m0 * K + kk, K);
    v16bf a1 = load_a(A + (size_t)(m0 + 16) * K + kk, K);
    v16bf b0 = load_b(Wb + (size_t)n0 * K + kk, K);
    v16bf b1 = load_b(Wb + (size_t)(n0 + 16) * K + kk, K);
    acc[0][0] = wmma_bf(a0, b0, acc[0][0]);
    acc[0][1] = wmma_bf(a0, b1, acc[0][1]);
    acc[1][0] = wmma_bf(a1, b0, acc[1][0]);
    acc[1][1] = wmma_bf(a1, b1, acc[1][1]);
  }
  int nlo = l & 15, mhi = (l >> 4) * 8;
  for (int mt = 0; mt < 2; ++mt)
    for (int nt = 0; nt < 2; ++nt)
#pragma unroll
      for (int r = 0; r < 8; ++r) {
        int tok = m0 + mt * 16 + mhi + r;
        int f = n0 + nt * 16 + nlo;
        out[(size_t)tok * N + f] = acc[mt][nt][r] + bias[f];
      }
}

// ---------- host ----------
extern "C" void kernel_launch(void* const* d_in, const int* in_sizes, int n_in,
                              void* d_out, int out_size, void* d_ws, size_t ws_size,
                              hipStream_t stream) {
  const float* x         = (const float*)d_in[0];
  const float* mask      = (const float*)d_in[1];
  const float* x_text    = (const float*)d_in[2];
  const float* mask_text = (const float*)d_in[3];
  const float* Wqkv      = (const float*)d_in[4];
  const float* bqkv      = (const float*)d_in[5];
  const float* Wproj     = (const float*)d_in[6];
  const float* bproj     = (const float*)d_in[7];
  const float* rpb       = (const float*)d_in[8];
  const float* Wqkv_t    = (const float*)d_in[9];
  const float* bqkv_t    = (const float*)d_in[10];
  const float* Wproj_t   = (const float*)d_in[11];
  const float* bproj_t   = (const float*)d_in[12];
  const float* i2t       = (const float*)d_in[13];
  const float* t2t       = (const float*)d_in[14];

  u16* w = (u16*)d_ws;
  size_t off = 0;
  auto take = [&](size_t n) { u16* p = w + off; off += n; return p; };
  u16* xb       = take(12845056);   // x bf16 [50176][256]
  u16* xtb      = take(524288);     // x_text bf16 [1024][512]
  u16* Wqkv_b   = take(196608);
  u16* Wqkv_t_b = take(393216);
  u16* Wproj_b  = take(65536);
  u16* Wproj_t_b= take(131072);
  u16* q        = take(16777216);   // [1024*8][64][32]
  u16* k        = take(16777216);
  u16* vT       = take(16777216);   // [1024*8][32][64]
  u16* qt       = take(262144);     // [16*8][64][32]
  u16* kt       = take(262144);
  u16* vtT      = take(262144);     // [16*8][32][64]
  u16* attn_out = take(12845056);   // [50176][256]
  u16* out_text = take(262144);     // [1024][256]

  float* x_out      = (float*)d_out;
  float* x_text_out = x_out + 12845056;

  k_cvt<<<4096, 256, 0, stream>>>(x, xb, 12845056);
  k_cvt<<<512,  256, 0, stream>>>(x_text, xtb, 524288);
  k_cvt<<<192,  256, 0, stream>>>(Wqkv, Wqkv_b, 196608);
  k_cvt<<<384,  256, 0, stream>>>(Wqkv_t, Wqkv_t_b, 393216);
  k_cvt<<<64,   256, 0, stream>>>(Wproj, Wproj_b, 65536);
  k_cvt<<<128,  256, 0, stream>>>(Wproj_t, Wproj_t_b, 131072);
  k_zero_pads<<<8192, 512, 0, stream>>>(q, k, vT);

  k_qkv_img<<<392 * 12, 256, 0, stream>>>(xb, Wqkv_b, bqkv, q, k, vT);
  k_qkv_txt<<<8 * 12,   256, 0, stream>>>(xtb, Wqkv_t_b, bqkv_t, qt, kt, vtT);

  k_attn_img<<<8192, 256, 0, stream>>>(q, k, vT, kt, vtT, rpb, mask, mask_text, i2t, attn_out);
  k_attn_text<<<128, 128, 0, stream>>>(qt, k, vT, kt, vtT, i2t, t2t, mask_text, out_text);

  k_proj<<<392 * 4, 256, 0, stream>>>(attn_out, Wproj_b, bproj, x_out, 256, 256, 4);
  k_proj<<<8 * 8,   256, 0, stream>>>(out_text, Wproj_t_b, bproj_t, x_text_out, 512, 256, 8);
}
